// MNIST_SNN_41626823033450
// MI455X (gfx1250) — matrix operand order repaired and checked by reading, not compiled, not run on hardware
//
#include <hip/hip_runtime.h>
#include <stdint.h>

typedef __attribute__((ext_vector_type(16))) _Float16 v16h;
typedef __attribute__((ext_vector_type(8)))  _Float16 v8h;
typedef __attribute__((ext_vector_type(8)))  float    v8f;

#define B_SZ   256
#define T_LEN  256
#define C0     784
#define C1     128
#define C2     64
#define C3     10

// LDS byte offsets (dynamic shared memory; total 213056 B <= 320KB/WGP)
#define OFF_STAGE 0        // double-buffered x slab: 2 x (32 rows x 256 f32) = 65536
#define OFF_XT    65536    // double-buffered xT[t][k]: 2 x (256 x 32 f16)   = 32768
#define OFF_H1T   98304    // z1T/h1T [256][136] f16 (in-place scan)         = 69632
#define OFF_H2T   167936   // z2T/h2T [256][72]  f16 (in-place scan)         = 36864
#define OFF_Z3    204800   // z3 [16][258] f16 (in-place scan)               = 8256
#define SMEM_TOTAL 213056

#define P_H1  136
#define P_H2  72
#define P_Z3  258

// ---- CDNA5 async global->LDS copy (ASYNCcnt-tracked) --------------------
__device__ __forceinline__ void async_load_b128(uint32_t lds_off, const void* gptr) {
    asm volatile("global_load_async_to_lds_b128 %0, %1, off"
                 :: "v"(lds_off), "v"((uint64_t)(uintptr_t)gptr)
                 : "memory");
}
__device__ __forceinline__ void wait_async0() {
    asm volatile("s_wait_asynccnt 0" ::: "memory");
}

__device__ __forceinline__ v16h load_bfrag(const _Float16* p) {
    v8h b0 = *(const v8h*)p;          // 16B aligned at all call sites
    v8h b1 = *(const v8h*)(p + 8);
    return __builtin_shufflevector(b0, b1, 0,1,2,3,4,5,6,7,8,9,10,11,12,13,14,15);
}

__device__ __forceinline__ v8f wmma16(const v16h& a, const v16h& b, const v8f& c) {
    return __builtin_amdgcn_wmma_f32_16x16x32_f16(false, a, false, b, (short)0, c, false, false);
}

__device__ __forceinline__ uint32_t pk2(float a, float b) {
    union { _Float16 h[2]; uint32_t u; } u;
    u.h[0] = (_Float16)a; u.h[1] = (_Float16)b;
    return u.u;                        // -> v_cvt_pk_f16_f32
}

// Wave-local transpose: wave w converts its own slab rows k=4w..4w+3 into
// xT[t][k] f16 columns (zero-fill past the K tail). No cross-wave data.
__device__ __forceinline__ void transpose_slab(_Float16* __restrict__ xTn,
                                               const float* __restrict__ sl,
                                               int wave, int lane, int nrows) {
    const int k0 = 4 * wave;
    if (k0 < nrows) {
        #pragma unroll
        for (int m = 0; m < 8; ++m) {
            const int t = lane + 32 * m;
            const float* sp = sl + k0 * 256 + t;
            const uint32_t lo = pk2(sp[0],   sp[256]);
            const uint32_t hi = pk2(sp[512], sp[768]);
            *(uint64_t*)(xTn + t * 32 + k0) = (uint64_t)lo | ((uint64_t)hi << 32);
        }
    } else {
        #pragma unroll
        for (int m = 0; m < 8; ++m)
            *(uint64_t*)(xTn + (lane + 32 * m) * 32 + k0) = 0ull;
    }
}

extern "C" __global__ __launch_bounds__(256)
void snn_fused(const float* __restrict__ x,  const float* __restrict__ W1,
               const float* __restrict__ W2, const float* __restrict__ W3,
               const float* __restrict__ cdec, const float* __restrict__ vdec,
               float* __restrict__ out)
{
    extern __shared__ char smem[];
    float*    stagef = (float*)   (smem + OFF_STAGE);   // [2][32][256] f32
    _Float16* xTbuf  = (_Float16*)(smem + OFF_XT);      // [2][256][32] f16
    _Float16* h1T    = (_Float16*)(smem + OFF_H1T);     // [256][136]   f16
    _Float16* h2T    = (_Float16*)(smem + OFF_H2T);     // [256][72]    f16
    _Float16* z3     = (_Float16*)(smem + OFF_Z3);      // [16][258]    f16

    const int b    = blockIdx.x;
    const int tid  = threadIdx.x;
    const int wave = tid >> 5;
    const int lane = tid & 31;
    const int lmod = lane & 15;   // M/N position within 16-wide tile
    const int hk   = lane >> 4;   // lane-half selector for K layout

    const char* xb = (const char*)(x + (size_t)b * C0 * T_LEN);

    // ====================== Layer 1: W1(128x784) x x[b](784x256) ==========
    // Wave tiling: wave = (M-pair mp, N-half nh); each B-frag feeds 2 WMMAs.
    const int mp = wave & 3;      // M tiles 2mp, 2mp+1 (rows 32mp..32mp+31)
    const int nh = wave >> 2;     // N tiles nh*8 .. nh*8+7

    // Prologue: wave-local async load + transpose of slab 0 -> xT[0]
    {
        const char*    src = xb + (size_t)wave * 4096;
        const uint32_t dst = (uint32_t)(OFF_STAGE + wave * 4096);
        #pragma unroll
        for (int j = 0; j < 8; ++j)
            async_load_b128(dst + lane * 16 + j * 512, src + lane * 16 + j * 512);
        wait_async0();
        transpose_slab(xTbuf, stagef, wave, lane, 32);
    }

    const float* w1rowA = W1 + (32 * mp + lmod) * C0;  // M = 32mp + lane%16
    const float* w1rowB = w1rowA + 16 * C0;            // M = 32mp + 16 + lane%16

    // A fragments for ks=0 (contiguous K layout: half h holds K=h*16+0..15)
    v16h aA, aB;
    #pragma unroll
    for (int j = 0; j < 8; ++j) {
        aA[j]     = (_Float16)w1rowA[hk * 16 + j];
        aA[8 + j] = (_Float16)w1rowA[hk * 16 + 8 + j];
        aB[j]     = (_Float16)w1rowB[hk * 16 + j];
        aB[8 + j] = (_Float16)w1rowB[hk * 16 + 8 + j];
    }
    __syncthreads();

    v8f acc[2][8] = {};
    for (int ks = 0; ks < 25; ++ks) {
        const int  cur = ks & 1, nxt = (ks + 1) & 1;
        const bool haveNext = (ks + 1 < 25);
        const int  nrowsN   = ((ks + 1) == 24) ? 16 : 32;
        // Issue wave-local async load of next slab (overlaps WMMA below)
        if (haveNext && 4 * wave < nrowsN) {
            const char*    src = xb + (size_t)(ks + 1) * 32768 + (size_t)wave * 4096;
            const uint32_t dst = (uint32_t)(OFF_STAGE + nxt * 32768 + wave * 4096);
            #pragma unroll
            for (int j = 0; j < 8; ++j)
                async_load_b128(dst + lane * 16 + j * 512, src + lane * 16 + j * 512);
        }
        // Issue raw A loads for ks+1 now; convert after the WMMA loop (latency
        // hidden behind 16 WMMAs + barrier).
        float arA[16], arB[16];
        if (haveNext) {
            const int  kb   = (ks + 1) * 32;
            const bool full = (ks + 1) < 24;   // tail step: K 784..799 are zero
            #pragma unroll
            for (int j = 0; j < 8; ++j) {
                const bool ok = full || hk == 0;
                arA[j]     = ok ? w1rowA[kb + hk * 16 + j] : 0.0f;
                arB[j]     = ok ? w1rowB[kb + hk * 16 + j] : 0.0f;
                arA[8 + j] = ok ? w1rowA[kb + hk * 16 + 8 + j] : 0.0f;
                arB[8 + j] = ok ? w1rowB[kb + hk * 16 + 8 + j] : 0.0f;
            }
        }
        // 16 WMMAs (2 M-tiles x 8 N-tiles); B-frags pipelined 2 tiles ahead
        const _Float16* bbase = xTbuf + cur * 8192 + (nh * 128 + lmod) * 32 + hk * 16;
        v16h bf0 = load_bfrag(bbase);
        v16h bf1 = load_bfrag(bbase + 512);
        #pragma unroll
        for (int nt = 0; nt < 6; ++nt) {
            v16h bn = load_bfrag(bbase + (nt + 2) * 512);
            acc[0][nt] = wmma16(aA, bf0, acc[0][nt]);
            acc[1][nt] = wmma16(aB, bf0, acc[1][nt]);
            bf0 = bf1; bf1 = bn;
        }
        acc[0][6] = wmma16(aA, bf0, acc[0][6]);
        acc[1][6] = wmma16(aB, bf0, acc[1][6]);
        acc[0][7] = wmma16(aA, bf1, acc[0][7]);
        acc[1][7] = wmma16(aB, bf1, acc[1][7]);
        if (haveNext) {
            // convert prefetched A for next step (loadcnt wait lands here)
            #pragma unroll
            for (int j = 0; j < 16; ++j) {
                aA[j] = (_Float16)arA[j];
                aB[j] = (_Float16)arB[j];
            }
            // Wave-local: wait own async, transpose own rows into xT[nxt]
            wait_async0();
            transpose_slab(xTbuf + nxt * 8192, stagef + nxt * 8192, wave, lane, nrowsN);
        }
        __syncthreads();   // publish xT[nxt] for next compute step
    }
    // Store z1 transposed [t][o]: D rows M=r+8*hk are contiguous -> one b128/tile
    #pragma unroll
    for (int p = 0; p < 2; ++p) {
        #pragma unroll
        for (int nt = 0; nt < 8; ++nt) {
            const int t = (nh * 8 + nt) * 16 + lmod;
            v8h zh;
            #pragma unroll
            for (int r = 0; r < 8; ++r) zh[r] = (_Float16)acc[p][nt][r];
            *(v8h*)(h1T + t * P_H1 + 16 * (2 * mp + p) + 8 * hk) = zh;
        }
    }
    __syncthreads();

    // LIF scan layer 1 (in place: z slot becomes spike after consumption)
    if (tid < C1) {
        const float a1 = 1.0f - cdec[0], a2 = 1.0f - vdec[0];
        float cur = 0.0f, vol = 0.0f;
        for (int t = 0; t < T_LEN; ++t) {
            cur = a1 * cur + (float)h1T[t * P_H1 + tid];
            vol = a2 * vol + cur;
            const bool sp = (vol >= 1.0f);
            h1T[t * P_H1 + tid] = (_Float16)(sp ? 1.0f : 0.0f);
            vol = sp ? 0.0f : vol;
        }
    }
    __syncthreads();

    // ====================== Layer 2: W2(64x128) x h1(128x256) =============
    {
        const int mp2 = wave & 1;   // M tiles 2mp2, 2mp2+1 (rows 32mp2..)
        const int nq  = wave >> 1;  // N tiles nq*4 .. nq*4+3
        const float* w2rowA = W2 + (32 * mp2 + lmod) * C1;
        const float* w2rowB = w2rowA + 16 * C1;
        v8f acc2[2][4] = {};
        #pragma unroll
        for (int ks = 0; ks < 4; ++ks) {
            v16h a2A, a2B;
            #pragma unroll
            for (int j = 0; j < 8; ++j) {
                a2A[j]     = (_Float16)w2rowA[ks * 32 + hk * 16 + j];
                a2A[8 + j] = (_Float16)w2rowA[ks * 32 + hk * 16 + 8 + j];
                a2B[j]     = (_Float16)w2rowB[ks * 32 + hk * 16 + j];
                a2B[8 + j] = (_Float16)w2rowB[ks * 32 + hk * 16 + 8 + j];
            }
            const _Float16* bb = h1T + (nq * 64 + lmod) * P_H1 + ks * 32 + hk * 16;
            v16h bf0 = load_bfrag(bb);
            v16h bf1 = load_bfrag(bb + 16 * P_H1);
            #pragma unroll
            for (int nt = 0; nt < 2; ++nt) {
                v16h bn = load_bfrag(bb + (nt + 2) * (16 * P_H1));
                acc2[0][nt] = wmma16(a2A, bf0, acc2[0][nt]);
                acc2[1][nt] = wmma16(a2B, bf0, acc2[1][nt]);
                bf0 = bf1; bf1 = bn;
            }
            acc2[0][2] = wmma16(a2A, bf0, acc2[0][2]);
            acc2[1][2] = wmma16(a2B, bf0, acc2[1][2]);
            acc2[0][3] = wmma16(a2A, bf1, acc2[0][3]);
            acc2[1][3] = wmma16(a2B, bf1, acc2[1][3]);
        }
        #pragma unroll
        for (int p = 0; p < 2; ++p) {
            #pragma unroll
            for (int nt = 0; nt < 4; ++nt) {
                const int t = (nq * 4 + nt) * 16 + lmod;
                v8h zh;
                #pragma unroll
                for (int r = 0; r < 8; ++r) zh[r] = (_Float16)acc2[p][nt][r];
                *(v8h*)(h2T + t * P_H2 + 16 * (2 * mp2 + p) + 8 * hk) = zh;
            }
        }
    }
    __syncthreads();

    if (tid < C2) {
        const float a1 = 1.0f - cdec[1], a2 = 1.0f - vdec[1];
        float cur = 0.0f, vol = 0.0f;
        for (int t = 0; t < T_LEN; ++t) {
            cur = a1 * cur + (float)h2T[t * P_H2 + tid];
            vol = a2 * vol + cur;
            const bool sp = (vol >= 1.0f);
            h2T[t * P_H2 + tid] = (_Float16)(sp ? 1.0f : 0.0f);
            vol = sp ? 0.0f : vol;
        }
    }
    __syncthreads();

    // ====================== Layer 3: W3(10x64) x h2(64x256) ===============
    {
        const bool valid = (lmod < C3);
        v8f acc3[2] = {};
        #pragma unroll
        for (int ks = 0; ks < 2; ++ks) {
            v16h a;
            #pragma unroll
            for (int j = 0; j < 8; ++j) {
                float w0 = valid ? W3[lmod * C2 + ks * 32 + hk * 16 + j]     : 0.0f;
                float w1 = valid ? W3[lmod * C2 + ks * 32 + hk * 16 + 8 + j] : 0.0f;
                a[j] = (_Float16)w0;  a[8 + j] = (_Float16)w1;
            }
            const _Float16* bb = h2T + (wave * 32 + lmod) * P_H2 + ks * 32 + hk * 16;
            v16h bf = load_bfrag(bb);
            v16h bn = load_bfrag(bb + 16 * P_H2);
            acc3[0] = wmma16(a, bf, acc3[0]);
            acc3[1] = wmma16(a, bn, acc3[1]);
        }
        #pragma unroll
        for (int nt = 0; nt < 2; ++nt) {
            const int t = (wave * 2 + nt) * 16 + lmod;
            #pragma unroll
            for (int r = 0; r < 8; ++r)
                z3[(r + 8 * hk) * P_Z3 + t] = (_Float16)acc3[nt][r];
        }
    }
    __syncthreads();

    // LIF scan layer 3 (in place)
    if (tid < C3) {
        const float a1 = 1.0f - cdec[2], a2 = 1.0f - vdec[2];
        float cur = 0.0f, vol = 0.0f;
        for (int t = 0; t < T_LEN; ++t) {
            cur = a1 * cur + (float)z3[tid * P_Z3 + t];
            vol = a2 * vol + cur;
            const bool sp = (vol >= 1.0f);
            z3[tid * P_Z3 + t] = (_Float16)(sp ? 1.0f : 0.0f);
            vol = sp ? 0.0f : vol;
        }
    }
    __syncthreads();

    // Coalesced output: out[b][o][t] f32
    for (int i = tid; i < C3 * T_LEN; i += 256)
        out[(size_t)b * C3 * T_LEN + i] = (float)z3[(i >> 8) * P_Z3 + (i & 255)];
}

extern "C" void kernel_launch(void* const* d_in, const int* in_sizes, int n_in,
                              void* d_out, int out_size, void* d_ws, size_t ws_size,
                              hipStream_t stream) {
    (void)in_sizes; (void)n_in; (void)d_ws; (void)ws_size; (void)out_size;
    const float* x  = (const float*)d_in[0];
    const float* W1 = (const float*)d_in[1];
    const float* W2 = (const float*)d_in[2];
    const float* W3 = (const float*)d_in[3];
    const float* cd = (const float*)d_in[4];
    const float* vd = (const float*)d_in[5];
    float* out = (float*)d_out;

    // >64KB dynamic LDS opt-in (idempotent; safe under graph capture)
    hipFuncSetAttribute(reinterpret_cast<const void*>(snn_fused),
                        hipFuncAttributeMaxDynamicSharedMemorySize, SMEM_TOTAL);

    snn_fused<<<dim3(B_SZ), dim3(256), SMEM_TOTAL, stream>>>(x, W1, W2, W3, cd, vd, out);
}
